// ConvAttnPool_60370060312691
// MI455X (gfx1250) — matrix operand order repaired and checked by reading, not compiled, not run on hardware
//
#include <hip/hip_runtime.h>
#include <math.h>

#define BB 8
#define NN 2500
#define EE 100
#define FF 50
#define YY 8922
#define KW 9

#define NPAD 2512   // 157*16
#define YPAD 8928   // 558*16
#define FPAD 64
#define NT_N 157
#define NT_Y 558
#define SROW 2500   // LDS row stride (floats); 2500%64=4 -> conflict-free phase-3 reads, no pad

typedef __attribute__((ext_vector_type(16))) _Float16 v16h;
typedef __attribute__((ext_vector_type(8)))  _Float16 v8h;
typedef __attribute__((ext_vector_type(8)))  float    v8f;
typedef __attribute__((ext_vector_type(2)))  float    v2f;

// ---------------------------------------------------------------- U_w -> f16 padded [YPAD][64]; also zero loss slot
__global__ void k_prep_uw(const float* __restrict__ Uw, _Float16* __restrict__ uw16,
                          float* __restrict__ loss) {
    int idx = blockIdx.x * blockDim.x + threadIdx.x;
    if (idx == 0) *loss = 0.f;
    if (idx >= YPAD * FPAD) return;
    int y = idx >> 6, f = idx & 63;
    float v = (y < YY && f < FF) ? Uw[y * FF + f] : 0.f;
    uw16[idx] = (_Float16)v;
}

// ---------------------------------------------------------------- embed + conv1d + tanh
__global__ void __launch_bounds__(64) k_conv(const int* __restrict__ x,
                                             const float* __restrict__ embW,
                                             const float* __restrict__ convW,
                                             const float* __restrict__ convB,
                                             float* __restrict__ h32,
                                             _Float16* __restrict__ h16) {
    __shared__ float se[KW * EE];   // se[k*EE + e]
    int n = blockIdx.x;             // 0..NPAD-1
    int b = blockIdx.y;
    int tid = threadIdx.x;
    if (n < NN) {
        for (int i = tid; i < KW * EE; i += 64) {
            int k = i / EE, e = i - k * EE;
            int pos = n + k - KW / 2;
            float v = 0.f;
            if (pos >= 0 && pos < NN) {
                int tok = x[b * NN + pos];
                v = embW[(size_t)tok * EE + e];
            }
            se[i] = v;
        }
    }
    __syncthreads();
    int f = tid;
    float hv = 0.f;
    if (n < NN && f < FF) {
        float acc = convB[f];
        const float* wf = convW + (size_t)f * EE * KW;   // conv_w[f][e][k]
        for (int e = 0; e < EE; ++e) {
            const float* we = wf + e * KW;
            const float* le = se + e;
            #pragma unroll
            for (int k = 0; k < KW; ++k) acc += le[k * EE] * we[k];
        }
        hv = tanhf(acc);
        h32[((size_t)b * NN + n) * FF + f] = hv;
    }
    if (f < FPAD) h16[((size_t)b * NPAD + n) * FPAD + f] = (_Float16)hv;
}

// ---------------------------------------------------------------- fused scores -> softmax -> m
// one workgroup (8 waves) per (b, 16-row y tile); score stripe [16][2500] lives in LDS.
// LDS = 160,000 B -> 2 workgroups per WGP (320 KB).
__global__ void __launch_bounds__(256) k_fused(const _Float16* __restrict__ h16,
                                               const _Float16* __restrict__ uw16,
                                               const float* __restrict__ h32,
                                               float* __restrict__ alphaG,
                                               float* __restrict__ mws) {
    extern __shared__ float smem[];
    float* sA = smem;                    // [16][SROW] scores/alpha stripe

    int t  = blockIdx.x;
    int b  = t / NT_Y;
    int ty = t - b * NT_Y;
    int y0 = ty * 16;
    int lane = threadIdx.x & 31;
    int w    = threadIdx.x >> 5;
    int half = lane >> 4;
    int l    = lane & 15;

    // ---- phase 1: scores tile-row into LDS (f16 WMMA, K=64). A (U_w tile) loaded once.
    {
        const _Float16* arow = uw16 + (size_t)(y0 + l) * FPAD;
        v16h a0 = *(const v16h*)(arow + half * 16);
        v16h a1 = *(const v16h*)(arow + 32 + half * 16);
        for (int tn = w; tn < NT_N; tn += 8) {
            int n0 = tn * 16;
            const _Float16* brow = h16 + ((size_t)b * NPAD + (n0 + l)) * FPAD;
            v16h b0 = *(const v16h*)(brow + half * 16);
            v16h b1 = *(const v16h*)(brow + 32 + half * 16);
            v8f c = {};
            c = __builtin_amdgcn_wmma_f32_16x16x32_f16(false, a0, false, b0, (short)0, c, false, false);
            c = __builtin_amdgcn_wmma_f32_16x16x32_f16(false, a1, false, b1, (short)0, c, false, false);
            if (n0 + l < NN) {                 // last tile: cols 2500..2511 must not spill
                #pragma unroll
                for (int g = 0; g < 8; ++g)
                    sA[(g + half * 8) * SROW + n0 + l] = c[g];  // rows 8 apart -> bank offset 32
            }
        }
    }
    __syncthreads();

    // ---- phase 2: softmax per row in LDS; write normalized alpha to HBM once (non-temporal)
    #pragma unroll
    for (int rr = 0; rr < 2; ++rr) {
        int r = w * 2 + rr;
        float*  row  = sA + r * SROW;
        float4* row4 = (float4*)row;           // 625 groups, base 10,000B*r -> 16B aligned
        int y = y0 + r;
        float mx = -3.4e38f;
        for (int g = lane; g < NN / 4; g += 32) {
            float4 v = row4[g];
            mx = fmaxf(fmaxf(fmaxf(mx, v.x), v.y), fmaxf(v.z, v.w));
        }
        #pragma unroll
        for (int off = 16; off > 0; off >>= 1) mx = fmaxf(mx, __shfl_xor(mx, off, 32));
        float s = 0.f;
        for (int g = lane; g < NN / 4; g += 32) {
            float4 v = row4[g];
            v.x = expf(v.x - mx); v.y = expf(v.y - mx);
            v.z = expf(v.z - mx); v.w = expf(v.w - mx);
            row4[g] = v;
            s += (v.x + v.y) + (v.z + v.w);
        }
        #pragma unroll
        for (int off = 16; off > 0; off >>= 1) s += __shfl_xor(s, off, 32);
        float inv = 1.f / s;
        bool yok = (y < YY);
        float* gout = alphaG + ((size_t)b * YY + (yok ? y : 0)) * NN;
        for (int i = lane; i < NN; i += 32) {   // scalar: alpha region is only 4B-aligned;
            float v = row[i] * inv;             // lane-contiguous b32 NT stores = 128B lines
            row[i] = v;
            if (yok) __builtin_nontemporal_store(v, gout + i);
        }
    }
    __syncthreads();

    // ---- phase 3: m[16][F] = alpha(LDS) @ h (f32 WMMA 16x16x4); waves 0..3, one f-tile each, full K
    if (w < 4) {
        int f0 = w * 16;
        int f  = f0 + l;
        bool fok = (f < FF);
        v8f c = {};
        for (int k = 0; k < NN / 4; ++k) {
            int nb = k * 4 + half * 2;          // lane<16: K0,K1 ; lane>=16: K2,K3
            float2 av = *(const float2*)(sA + (size_t)l * SROW + nb);
            v2f a; a.x = av.x; a.y = av.y;
            const float* bp = h32 + ((size_t)b * NN + nb) * FF + f;
            v2f bm;
            bm.x = fok ? bp[0]  : 0.f;
            bm.y = fok ? bp[FF] : 0.f;
            c = __builtin_amdgcn_wmma_f32_16x16x4_f32(false, a, false, bm, (short)0, c, false, false);
        }
        #pragma unroll
        for (int g = 0; g < 8; ++g)
            mws[((size_t)b * YPAD + y0 + g + half * 8) * FPAD + f0 + l] = c[g];
    }
}

// ---------------------------------------------------------------- logits, sigmoid, BCE
__global__ void __launch_bounds__(256) k_logits(const float* __restrict__ mws,
                                                const float* __restrict__ finalW,
                                                const float* __restrict__ finalB,
                                                const float* __restrict__ target,
                                                float* __restrict__ yhat,
                                                float* __restrict__ loss) {
    __shared__ float red[256];
    int idx = blockIdx.x * blockDim.x + threadIdx.x;
    float term = 0.f;
    if (idx < BB * YY) {
        int b = idx / YY, y = idx - b * YY;
        const float* mrow = mws + ((size_t)b * YPAD + y) * FPAD;
        const float* frow = finalW + (size_t)y * FF;
        float acc = finalB[y];
        for (int f = 0; f < FF; ++f) acc += mrow[f] * frow[f];
        float p = 1.f / (1.f + expf(-acc));
        yhat[idx] = p;
        float pc = fminf(fmaxf(p, 1e-7f), 1.f - 1e-7f);
        float tg = target[idx];
        term = tg * logf(pc) + (1.f - tg) * log1pf(-pc);
    }
    red[threadIdx.x] = term;
    __syncthreads();
    for (int s = 128; s > 0; s >>= 1) {
        if (threadIdx.x < s) red[threadIdx.x] += red[threadIdx.x + s];
        __syncthreads();
    }
    if (threadIdx.x == 0) atomicAdd(loss, -red[0] / (float)(BB * YY));
}

extern "C" void kernel_launch(void* const* d_in, const int* in_sizes, int n_in,
                              void* d_out, int out_size, void* d_ws, size_t ws_size,
                              hipStream_t stream) {
    (void)in_sizes; (void)n_in; (void)out_size; (void)ws_size;
    const int*   x      = (const int*)  d_in[0];
    const float* target = (const float*)d_in[1];
    const float* embW   = (const float*)d_in[2];
    const float* convW  = (const float*)d_in[3];
    const float* convB  = (const float*)d_in[4];
    const float* Uw     = (const float*)d_in[5];
    const float* finalW = (const float*)d_in[6];
    const float* finalB = (const float*)d_in[7];

    float* out   = (float*)d_out;          // [yhat (B*Y) | loss (1) | alpha (B*Y*N)]
    float* yhat  = out;
    float* loss  = out + (size_t)BB * YY;
    float* alpha = loss + 1;

    char* ws = (char*)d_ws;
    float*    h32  = (float*)(ws);                 //  4,000,000 B
    _Float16* h16  = (_Float16*)(ws + 4000000);    //  2,572,288 B
    _Float16* uw16 = (_Float16*)(ws + 6572288);    //  1,142,784 B
    float*    mws  = (float*)(ws + 7715072);       // 18,284,544 B  (total ~26 MB)

    const size_t lds_bytes = (size_t)(16 * SROW) * sizeof(float);  // 160,000 B -> 2 WG/WGP

    hipLaunchKernelGGL(k_prep_uw, dim3((YPAD * FPAD + 255) / 256), dim3(256), 0, stream,
                       Uw, uw16, loss);
    hipLaunchKernelGGL(k_conv,    dim3(NPAD, BB),                  dim3(64),  0, stream,
                       x, embW, convW, convB, h32, h16);
    hipLaunchKernelGGL(k_fused,   dim3(BB * NT_Y),                 dim3(256), lds_bytes, stream,
                       h16, uw16, h32, alpha, mws);
    hipLaunchKernelGGL(k_logits,  dim3((BB * YY + 255) / 256),     dim3(256), 0, stream,
                       mws, finalW, finalB, target, yhat, loss);
}
